// MultiwayNetwork_61125974557706
// MI455X (gfx1250) — matrix operand order repaired and checked by reading, not compiled, not run on hardware
//
#include <hip/hip_runtime.h>
#include <hip/hip_bf16.h>

typedef __bf16 bf16_t;
typedef __attribute__((ext_vector_type(16))) __bf16 v16bf;
typedef __attribute__((ext_vector_type(8)))  __bf16 v8bf;
typedef __attribute__((ext_vector_type(8)))  float  v8f;

#define DM    2048
#define NTOK  16384          // 4 * 4096 tokens
#define KSTEP 32

// d_ws layout (int32 units): [0]=count route0, [1]=count route1,
// permA @ 16, permB @ 16+NTOK, then bf16 copies of W0|W1 @ WS_BF.
#define WS_PERMA 16
#define WS_PERMB (16 + NTOK)
#define WS_BF    (16 + 2 * NTOK)   // int offset; *4 bytes = 131,136 (16B aligned)

// ---------------------------------------------------------------------------
// Kernel 0: zero the partition counters (ws is poisoned / stale between calls)
// ---------------------------------------------------------------------------
__global__ void mw_zero_counters(int* ws) {
    if (threadIdx.x < 2) ws[threadIdx.x] = 0;
}

// ---------------------------------------------------------------------------
// Kernel 1: compact token ids by route. Slot order within a route is arbitrary
// (atomic), but the scattered output is deterministic.
// ---------------------------------------------------------------------------
__global__ void mw_partition(const long long* __restrict__ idx, int* __restrict__ ws) {
    int t = blockIdx.x * blockDim.x + threadIdx.x;
    if (t >= NTOK) return;
    if (idx[t] == 0) {
        int slot = atomicAdd(&ws[0], 1);
        ws[WS_PERMA + slot] = t;
    } else {
        int slot = atomicAdd(&ws[1], 1);
        ws[WS_PERMB + slot] = t;
    }
}

// ---------------------------------------------------------------------------
// Kernel 1b: one-time f32 -> bf16 conversion of W0|W1 into d_ws.
// ---------------------------------------------------------------------------
__global__ void mw_convert_w(const float* __restrict__ W0,
                             const float* __restrict__ W1,
                             v8bf* __restrict__ out) {
    const int i    = blockIdx.x * blockDim.x + threadIdx.x;   // [0, 2*DM*DM/8)
    const int half = DM * DM / 8;
    const float4* src = (i < half) ? (const float4*)W0 + (size_t)i * 2
                                   : (const float4*)W1 + (size_t)(i - half) * 2;
    const float4 f0 = src[0];
    const float4 f1 = src[1];
    v8bf o;
    o[0] = (bf16_t)f0.x; o[1] = (bf16_t)f0.y; o[2] = (bf16_t)f0.z; o[3] = (bf16_t)f0.w;
    o[4] = (bf16_t)f1.x; o[5] = (bf16_t)f1.y; o[6] = (bf16_t)f1.z; o[7] = (bf16_t)f1.w;
    out[i] = o;
}

// ---------------------------------------------------------------------------
// Shared per-wave tile setup. Returns false if this wave has no work.
// ---------------------------------------------------------------------------
struct TileInfo {
    const int* perm;
    int        route;   // 0 or 1
    int        cnt;     // valid rows in this 16-token tile
};

__device__ __forceinline__ bool tile_setup(const int* __restrict__ ws, int tile, TileInfo& ti) {
    const int n0cnt  = ws[0];
    const int n1cnt  = ws[1];
    const int tiles0 = (n0cnt + 15) >> 4;
    const int tiles1 = (n1cnt + 15) >> 4;
    if (tile < tiles0) {
        ti.perm  = ws + WS_PERMA + tile * 16;
        ti.route = 0;
        ti.cnt   = n0cnt - tile * 16;
    } else {
        const int t1 = tile - tiles0;
        if (t1 >= tiles1) return false;
        ti.perm  = ws + WS_PERMB + t1 * 16;
        ti.route = 1;
        ti.cnt   = n1cnt - t1 * 16;
    }
    if (ti.cnt > 16) ti.cnt = 16;
    return true;
}

// A-fragment load: f32 X row -> 16x32 bf16 fragment for this lane.
// Lane holds M = lane&15, K = half*16 + {0..15}.
__device__ __forceinline__ v16bf load_a_frag(const float* __restrict__ xk, int half) {
    const float4 f0 = *(const float4*)(xk + half * 16 + 0);
    const float4 f1 = *(const float4*)(xk + half * 16 + 4);
    const float4 f2 = *(const float4*)(xk + half * 16 + 8);
    const float4 f3 = *(const float4*)(xk + half * 16 + 12);
    v16bf a;
    a[0]  = (bf16_t)f0.x; a[1]  = (bf16_t)f0.y; a[2]  = (bf16_t)f0.z; a[3]  = (bf16_t)f0.w;
    a[4]  = (bf16_t)f1.x; a[5]  = (bf16_t)f1.y; a[6]  = (bf16_t)f1.z; a[7]  = (bf16_t)f1.w;
    a[8]  = (bf16_t)f2.x; a[9]  = (bf16_t)f2.y; a[10] = (bf16_t)f2.z; a[11] = (bf16_t)f2.w;
    a[12] = (bf16_t)f3.x; a[13] = (bf16_t)f3.y; a[14] = (bf16_t)f3.z; a[15] = (bf16_t)f3.w;
    return a;
}

// ---------------------------------------------------------------------------
// Kernel 2 (fast path): routed GEMM with pre-converted bf16 weights.
// Block = 8 waves covering 8 consecutive 16-token M-tiles; ALL waves share
// the same N-slice [blockIdx.y*128, +128), so their B-fragment loads are
// identical addresses -> one L2 fill per line, 7 WGP-cache hits. X + bf16 W
// together fit in the 192MB L2, so HBM traffic stays near the 300MB floor.
// ---------------------------------------------------------------------------
__global__ void __launch_bounds__(256)
mw_gemm_bf16w(const float* __restrict__ X,
              const bf16_t* __restrict__ Wbf,      // [2][DM][DM]
              const float* __restrict__ b0, const float* __restrict__ b1,
              const int*   __restrict__ ws,
              float*       __restrict__ out)
{
    const int lane = threadIdx.x & 31;
    const int wave = threadIdx.x >> 5;
    const int half = lane >> 4;
    const int ncol = lane & 15;

    TileInfo ti;
    if (!tile_setup(ws, blockIdx.x * 8 + wave, ti)) return;
    const float* __restrict__ bias = ti.route ? b1 : b0;

    const int mrow = lane & 15;
    const int tokA = (mrow < ti.cnt) ? ti.perm[mrow] : ti.perm[0];
    const float* __restrict__ xrow = X + tokA * DM;

    const int nwbase = blockIdx.y * 128;     // shared by all 8 waves

    // Single per-lane W base pointer; fragment rows reached via immediate
    // offsets t*16*DM elements (t*65536 bytes < 24-bit IOFFSET range).
    const bf16_t* __restrict__ wbase =
        Wbf + (size_t)ti.route * DM * DM + (size_t)(nwbase + ncol) * DM + half * 16;

    v8f acc[8];
#pragma unroll
    for (int t = 0; t < 8; ++t) { v8f z = {}; acc[t] = z; }

#pragma unroll 2
    for (int kb = 0; kb < DM; kb += KSTEP) {
        const v16bf a = load_a_frag(xrow + kb, half);
        if (kb + KSTEP < DM) __builtin_prefetch(xrow + kb + KSTEP + half * 8, 0, 3);

#pragma unroll
        for (int t = 0; t < 8; ++t) {
            const bf16_t* __restrict__ wrow = wbase + kb + t * (16 * DM);
            const v8bf p0 = *(const v8bf*)(wrow + 0);
            const v8bf p1 = *(const v8bf*)(wrow + 8);
            const v16bf b = __builtin_shufflevector(p0, p1,
                0, 1, 2, 3, 4, 5, 6, 7, 8, 9, 10, 11, 12, 13, 14, 15);
            acc[t] = __builtin_amdgcn_wmma_f32_16x16x32_bf16(
                false, a, false, b, (short)0, acc[t], false, false);
        }
    }

    // Epilogue: bias add + scatter. C/D layout: VGPR v -> row half*8+v, col ncol.
    int tokrow[8];
#pragma unroll
    for (int v = 0; v < 8; ++v) {
        const int m = half * 8 + v;
        tokrow[v] = (m < ti.cnt) ? ti.perm[m] : -1;
    }
#pragma unroll
    for (int t = 0; t < 8; ++t) {
        const int   n  = nwbase + t * 16 + ncol;
        const float bv = bias[n];
#pragma unroll
        for (int v = 0; v < 8; ++v) {
            if (tokrow[v] >= 0)
                out[tokrow[v] * DM + n] = acc[t][v] + bv;
        }
    }
}

// ---------------------------------------------------------------------------
// Kernel 2 (fallback): identical tiling, converts f32 W on the fly.
// Used only if d_ws is too small for the bf16 weight copies.
// ---------------------------------------------------------------------------
__global__ void __launch_bounds__(256)
mw_gemm_f32w(const float* __restrict__ X,
             const float* __restrict__ W0, const float* __restrict__ b0,
             const float* __restrict__ W1, const float* __restrict__ b1,
             const int*   __restrict__ ws,
             float*       __restrict__ out)
{
    const int lane = threadIdx.x & 31;
    const int wave = threadIdx.x >> 5;
    const int half = lane >> 4;
    const int ncol = lane & 15;

    TileInfo ti;
    if (!tile_setup(ws, blockIdx.x * 8 + wave, ti)) return;
    const float* __restrict__ bias = ti.route ? b1 : b0;

    const int mrow = lane & 15;
    const int tokA = (mrow < ti.cnt) ? ti.perm[mrow] : ti.perm[0];
    const float* __restrict__ xrow = X + tokA * DM;

    const int nwbase = blockIdx.y * 128;     // shared by all 8 waves

    // Single base pointer; row offsets t*16*DM floats (t*131072 B) fit IOFFSET.
    const float* __restrict__ wbase =
        (ti.route ? W1 : W0) + (size_t)(nwbase + ncol) * DM + half * 16;

    v8f acc[8];
#pragma unroll
    for (int t = 0; t < 8; ++t) { v8f z = {}; acc[t] = z; }

    for (int kb = 0; kb < DM; kb += KSTEP) {
        const v16bf a = load_a_frag(xrow + kb, half);

#pragma unroll
        for (int t = 0; t < 8; ++t) {
            const float* __restrict__ wrow = wbase + kb + t * (16 * DM);
            const float4 f0 = *(const float4*)(wrow + 0);
            const float4 f1 = *(const float4*)(wrow + 4);
            const float4 f2 = *(const float4*)(wrow + 8);
            const float4 f3 = *(const float4*)(wrow + 12);
            v16bf b;
            b[0]  = (bf16_t)f0.x; b[1]  = (bf16_t)f0.y; b[2]  = (bf16_t)f0.z; b[3]  = (bf16_t)f0.w;
            b[4]  = (bf16_t)f1.x; b[5]  = (bf16_t)f1.y; b[6]  = (bf16_t)f1.z; b[7]  = (bf16_t)f1.w;
            b[8]  = (bf16_t)f2.x; b[9]  = (bf16_t)f2.y; b[10] = (bf16_t)f2.z; b[11] = (bf16_t)f2.w;
            b[12] = (bf16_t)f3.x; b[13] = (bf16_t)f3.y; b[14] = (bf16_t)f3.z; b[15] = (bf16_t)f3.w;
            acc[t] = __builtin_amdgcn_wmma_f32_16x16x32_bf16(
                false, a, false, b, (short)0, acc[t], false, false);
        }
    }

    int tokrow[8];
#pragma unroll
    for (int v = 0; v < 8; ++v) {
        const int m = half * 8 + v;
        tokrow[v] = (m < ti.cnt) ? ti.perm[m] : -1;
    }
#pragma unroll
    for (int t = 0; t < 8; ++t) {
        const int   n  = nwbase + t * 16 + ncol;
        const float bv = bias[n];
#pragma unroll
        for (int v = 0; v < 8; ++v) {
            if (tokrow[v] >= 0)
                out[tokrow[v] * DM + n] = acc[t][v] + bv;
        }
    }
}

// ---------------------------------------------------------------------------
extern "C" void kernel_launch(void* const* d_in, const int* in_sizes, int n_in,
                              void* d_out, int out_size, void* d_ws, size_t ws_size,
                              hipStream_t stream) {
    const float*     X   = (const float*)d_in[0];
    const long long* idx = (const long long*)d_in[1];
    const float*     W0  = (const float*)d_in[2];
    const float*     b0  = (const float*)d_in[3];
    const float*     W1  = (const float*)d_in[4];
    const float*     b1  = (const float*)d_in[5];
    float*           out = (float*)d_out;
    int*             ws  = (int*)d_ws;

    mw_zero_counters<<<1, 32, 0, stream>>>(ws);
    mw_partition<<<NTOK / 256, 256, 0, stream>>>(idx, ws);

    // Worst case M tiles: ceil(n0/16) + ceil(n1/16) <= 1025 -> 129 groups of 8.
    // blockIdx.y: 16 N-slices of 128 columns (shared by all waves of a block).
    const dim3 grid(129, 16, 1);

    const size_t ws_need = (size_t)WS_BF * 4 + (size_t)2 * DM * DM * sizeof(bf16_t);
    if (ws_size >= ws_need) {
        bf16_t* Wbf = (bf16_t*)(ws + WS_BF);
        mw_convert_w<<<(2 * DM * DM / 8) / 256, 256, 0, stream>>>(W0, W1, (v8bf*)Wbf);
        mw_gemm_bf16w<<<grid, 256, 0, stream>>>(X, Wbf, b0, b1, ws, out);
    } else {
        mw_gemm_f32w<<<grid, 256, 0, stream>>>(X, W0, b0, W1, b1, ws, out);
    }
}
